// AERIALAgent_3410204033722
// MI455X (gfx1250) — compile-verified, matrix operand order checked by reading, not run on hardware
//
#include <hip/hip_runtime.h>
#include <hip/hip_bf16.h>

// ---------------------------------------------------------------------------
// AERIAL agent step, fused for gfx1250 (MI455X): all GEMMs via
// V_WMMA_F32_16X16X32_F16 (f16 inputs, fp32 accumulate). One wave32 per
// workgroup processes a 16-row batch tile end-to-end using private LDS.
// Weights are pre-converted to f16 into d_ws by pack_weights (L2-resident).
// ---------------------------------------------------------------------------

typedef __attribute__((ext_vector_type(16))) _Float16 v16h;
typedef __attribute__((ext_vector_type(8)))  _Float16 v8h;
typedef __attribute__((ext_vector_type(8)))  float    v8f;

#define B_TOT 65536
#define OBS_D 128
#define E_D   64
#define H_D   128
#define NT_D  7
#define A_D   16

// f16 weight workspace offsets (in halves), all 16B-aligned
#define WOFF_W1    0        // [64][128]
#define WOFF_W2    8192     // [64][64]
#define WOFF_WP    12288    // [64][128]
#define WOFF_INPJ  20480    // [192][64]  (rows 0..63 Wq, 64..127 Wk, 128..191 Wv)
#define WOFF_OUTPJ 32768    // [64][64]
#define WOFF_WRZ   36864    // [256][256] rows: 0..127 r-gate, 128..255 z-gate;
                            //            cols: 0..127 = wih, 128..255 = whh
#define WOFF_WNI   102400   // [128][128] = wih rows 256..383 (n-gate, input part)
#define WOFF_WNH   118784   // [128][128] = whh rows 256..383 (n-gate, hidden part)
#define WOFF_WPOL  135168   // [16][128]
#define W_TOTAL    137216

__device__ __forceinline__ v8f wmma16(v16h a, v16h b, v8f c) {
  // D = A(16x32 f16) x B(32x16 f16) + C(16x16 f32)
  return __builtin_amdgcn_wmma_f32_16x16x32_f16(false, a, false, b, (short)0, c,
                                                false, false);
}

// Fragment gather per ISA 7.12.2: lane l holds row/col (l&15); halves [0..7]
// are K = kbase + (l>>4)*8 .. +7, halves [8..15] are K = kbase+16+(l>>4)*8 .. +7.
// Both slices are contiguous -> two b128 loads.
__device__ __forceinline__ v16h frag16(const _Float16* p0) {
  v8h lo = *(const v8h*)p0;
  v8h hi = *(const v8h*)(p0 + 16);
  v16h r;
#pragma unroll
  for (int i = 0; i < 8; ++i) { r[i] = lo[i]; r[i + 8] = hi[i]; }
  return r;
}

// A-fragment from a row-major f16 LDS tile (rows = batch rows)
__device__ __forceinline__ v16h afrag_l(const _Float16* buf, int strideH,
                                        int kbase, int lane) {
  int m = lane & 15, hi = lane >> 4;
  return frag16(buf + m * strideH + kbase + hi * 8);
}

// B-fragment from row-major f16 weights W[N][K] (acts as B = W^T, lane = out col)
__device__ __forceinline__ v16h bfrag_w(const _Float16* W, int strideH, int n0,
                                        int kbase, int lane) {
  int n = lane & 15, hi = lane >> 4;
  return frag16(W + (n0 + n) * strideH + kbase + hi * 8);
}

// A-fragment gathered from row-major f32 global memory (cvt f32->f16)
__device__ __forceinline__ v16h afrag_g32(const float* base, int strideF,
                                          int kbase, int lane) {
  int m = lane & 15, hi = lane >> 4;
  const float* p = base + (size_t)m * strideF + kbase + hi * 8;
  v16h r;
#pragma unroll
  for (int i = 0; i < 8; ++i) {
    r[i]     = (_Float16)p[i];
    r[i + 8] = (_Float16)p[16 + i];
  }
  return r;
}

// Store 16x16 D tile (lane = col n, VGPR r = row) into f16 LDS, + bias, opt relu
__device__ __forceinline__ void store_tile(v8f acc, const float* bias, int bias0,
                                           _Float16* buf, int strideH, int ncol0,
                                           int lane, bool relu) {
  int n = lane & 15, mb = (lane >> 4) * 8;
  float bv = bias[bias0 + n];
#pragma unroll
  for (int r = 0; r < 8; ++r) {
    float v = acc[r] + bv;
    if (relu) v = fmaxf(v, 0.f);
    buf[(mb + r) * strideH + ncol0 + n] = (_Float16)v;
  }
}

// ---------------------------------------------------------------------------
// Weight conversion / packing (one-shot, tiny)
// ---------------------------------------------------------------------------
__global__ void pack_weights(const float* __restrict__ W1, const float* __restrict__ W2,
                             const float* __restrict__ Wp, const float* __restrict__ inw,
                             const float* __restrict__ outw, const float* __restrict__ wih,
                             const float* __restrict__ whh, const float* __restrict__ Wpol,
                             _Float16* __restrict__ ws) {
  int i = blockIdx.x * blockDim.x + threadIdx.x;
  if (i >= W_TOTAL) return;
  float v;
  if (i < WOFF_W2)              v = W1[i - WOFF_W1];
  else if (i < WOFF_WP)         v = W2[i - WOFF_W2];
  else if (i < WOFF_INPJ)       v = Wp[i - WOFF_WP];
  else if (i < WOFF_OUTPJ)      v = inw[i - WOFF_INPJ];
  else if (i < WOFF_WRZ)        v = outw[i - WOFF_OUTPJ];
  else if (i < WOFF_WNI) {      // [wih | whh] for r,z gates
    int j = i - WOFF_WRZ, row = j >> 8, col = j & 255;
    v = (col < 128) ? wih[row * 128 + col] : whh[row * 128 + (col - 128)];
  } else if (i < WOFF_WNH) {
    int j = i - WOFF_WNI;
    v = wih[(256 + (j >> 7)) * 128 + (j & 127)];
  } else if (i < WOFF_WPOL) {
    int j = i - WOFF_WNH;
    v = whh[(256 + (j >> 7)) * 128 + (j & 127)];
  } else {
    v = Wpol[i - WOFF_WPOL];
  }
  ws[i] = (_Float16)v;
}

// ---------------------------------------------------------------------------
// Fused forward kernel: one wave32 per block, 16 batch rows per wave.
// ---------------------------------------------------------------------------
__global__ __launch_bounds__(32) void aerial_fused(
    const float* __restrict__ obs, const float* __restrict__ pbel,
    const float* __restrict__ tmh,
    const float* __restrict__ b1, const float* __restrict__ b2,
    const float* __restrict__ bp, const float* __restrict__ in_b,
    const float* __restrict__ out_b, const float* __restrict__ bih,
    const float* __restrict__ bhh, const float* __restrict__ bpol,
    const _Float16* __restrict__ wf,
    float* __restrict__ out_logits, float* __restrict__ out_belief,
    float* __restrict__ out_attn)
{
  // f16 staging, +8-half row padding to spread LDS banks; all regions 16B aligned
  __shared__ __align__(16) _Float16 smem[26240];       // 52480 bytes
  _Float16* XCAT = smem;                 // 16 x 264: [h_obs | context | belief]
  _Float16* H1   = smem + 4224;          // 16 x 72
  _Float16* QIN  = smem + 5376;          // 16 x 72
  _Float16* QB   = smem + 6528;          // 16 x 72  (q, later reused for ctx)
  _Float16* KV   = smem + 7680;          // 7 x 16 x 72 (kv_in, later V)
  _Float16* KB   = smem + 15744;         // 7 x 16 x 72 (K)
  _Float16* BEL2 = smem + 23808;         // 16 x 136 (new belief, f16)
  float*    ATW  = (float*)(smem + 25984); // 16 x 8 f32 attn-weight accum

  const int lane = (int)threadIdx.x;     // 0..31, wave32
  const int row0 = (int)blockIdx.x * 16;

  for (int i = lane; i < 128; i += 32) ATW[i] = 0.f;

  // ---- 1) h1 = relu(obs @ W1^T + b1)                           [16x128]->[16x64]
  {
    v16h a[4];
#pragma unroll
    for (int kc = 0; kc < 4; ++kc)
      a[kc] = afrag_g32(obs + (size_t)row0 * OBS_D, OBS_D, kc * 32, lane);
#pragma unroll
    for (int nt = 0; nt < 4; ++nt) {
      v8f acc = {};
#pragma unroll
      for (int kc = 0; kc < 4; ++kc)
        acc = wmma16(a[kc], bfrag_w(wf + WOFF_W1, 128, nt * 16, kc * 32, lane), acc);
      store_tile(acc, b1, nt * 16, H1, 72, nt * 16, lane, true);
    }
  }
  // ---- 2) h_obs = relu(h1 @ W2^T + b2) -> XCAT[:,0:64]
  {
    v16h a[2];
#pragma unroll
    for (int kc = 0; kc < 2; ++kc) a[kc] = afrag_l(H1, 72, kc * 32, lane);
#pragma unroll
    for (int nt = 0; nt < 4; ++nt) {
      v8f acc = {};
#pragma unroll
      for (int kc = 0; kc < 2; ++kc)
        acc = wmma16(a[kc], bfrag_w(wf + WOFF_W2, 64, nt * 16, kc * 32, lane), acc);
      store_tile(acc, b2, nt * 16, XCAT, 264, nt * 16, lane, true);
    }
  }
  // ---- copy prev_belief (f32->f16) into XCAT[:,128:256]
  for (int i = lane; i < 16 * H_D; i += 32) {
    int r = i >> 7, c = i & 127;
    XCAT[r * 264 + 128 + c] = (_Float16)pbel[(size_t)(row0 + r) * H_D + c];
  }
  // ---- 3) q_in = belief @ Wp^T + bp
  {
    v16h a[4];
#pragma unroll
    for (int kc = 0; kc < 4; ++kc)
      a[kc] = afrag_g32(pbel + (size_t)row0 * H_D, H_D, kc * 32, lane);
#pragma unroll
    for (int nt = 0; nt < 4; ++nt) {
      v8f acc = {};
#pragma unroll
      for (int kc = 0; kc < 4; ++kc)
        acc = wmma16(a[kc], bfrag_w(wf + WOFF_WP, 128, nt * 16, kc * 32, lane), acc);
      store_tile(acc, bp, nt * 16, QIN, 72, nt * 16, lane, false);
    }
  }
  // ---- 4) q = q_in @ Wq^T + bq
  {
    v16h a[2];
#pragma unroll
    for (int kc = 0; kc < 2; ++kc) a[kc] = afrag_l(QIN, 72, kc * 32, lane);
#pragma unroll
    for (int nt = 0; nt < 4; ++nt) {
      v8f acc = {};
#pragma unroll
      for (int kc = 0; kc < 2; ++kc)
        acc = wmma16(a[kc], bfrag_w(wf + WOFF_INPJ, 64, nt * 16, kc * 32, lane), acc);
      store_tile(acc, in_b, nt * 16, QB, 72, nt * 16, lane, false);
    }
  }
  // ---- 5) per teammate: kv_in = tm @ Wp^T + bp; then K,V projections
  for (int t = 0; t < NT_D; ++t) {
    v16h a[4];
#pragma unroll
    for (int kc = 0; kc < 4; ++kc)
      a[kc] = afrag_g32(tmh + ((size_t)row0 * NT_D + t) * H_D, NT_D * H_D,
                        kc * 32, lane);
#pragma unroll
    for (int nt = 0; nt < 4; ++nt) {
      v8f acc = {};
#pragma unroll
      for (int kc = 0; kc < 4; ++kc)
        acc = wmma16(a[kc], bfrag_w(wf + WOFF_WP, 128, nt * 16, kc * 32, lane), acc);
      store_tile(acc, bp, nt * 16, KV + t * 1152, 72, nt * 16, lane, false);
    }
    v16h a2[2];                 // hoist kv A-frags so V may overwrite KV in place
#pragma unroll
    for (int kc = 0; kc < 2; ++kc) a2[kc] = afrag_l(KV + t * 1152, 72, kc * 32, lane);
#pragma unroll
    for (int nt = 0; nt < 4; ++nt) {
      v8f ak = {}, av = {};
#pragma unroll
      for (int kc = 0; kc < 2; ++kc) {
        ak = wmma16(a2[kc], bfrag_w(wf + WOFF_INPJ, 64, 64 + nt * 16, kc * 32, lane), ak);
        av = wmma16(a2[kc], bfrag_w(wf + WOFF_INPJ, 64, 128 + nt * 16, kc * 32, lane), av);
      }
      store_tile(ak, in_b, 64 + nt * 16,  KB + t * 1152, 72, nt * 16, lane, false);
      store_tile(av, in_b, 128 + nt * 16, KV + t * 1152, 72, nt * 16, lane, false);
    }
  }
  // ---- 6) attention: each lane handles two (row, head) pairs (4 heads x 16 rows)
#pragma unroll
  for (int pp = 0; pp < 2; ++pp) {
    int p = lane * 2 + pp;
    int row = p >> 2, head = p & 3;
    _Float16* qrow = QB + row * 72 + head * 16;
    float qv[16];
#pragma unroll
    for (int i = 0; i < 16; ++i) qv[i] = (float)qrow[i];
    float s[NT_D]; float mx = -1e30f;
    for (int t = 0; t < NT_D; ++t) {
      const _Float16* krow = KB + t * 1152 + row * 72 + head * 16;
      float acc = 0.f;
#pragma unroll
      for (int i = 0; i < 16; ++i) acc += qv[i] * (float)krow[i];
      s[t] = acc * 0.25f;                       // 1/sqrt(16)
      mx = fmaxf(mx, s[t]);
    }
    float sum = 0.f;
    for (int t = 0; t < NT_D; ++t) { s[t] = expf(s[t] - mx); sum += s[t]; }
    float inv = 1.f / sum;
    float ctx[16];
#pragma unroll
    for (int i = 0; i < 16; ++i) ctx[i] = 0.f;
    for (int t = 0; t < NT_D; ++t) {
      float w = s[t] * inv;
      const _Float16* vrow = KV + t * 1152 + row * 72 + head * 16;
#pragma unroll
      for (int i = 0; i < 16; ++i) ctx[i] += w * (float)vrow[i];
      atomicAdd(&ATW[row * 8 + t], 0.25f * w);  // mean over 4 heads
    }
    // reuse QB as ctx staging: even-pair writes never alias odd-pair q reads
#pragma unroll
    for (int i = 0; i < 16; ++i) qrow[i] = (_Float16)ctx[i];
  }
  // emit attention weights
  for (int i = lane; i < 16 * NT_D; i += 32) {
    int r = i / NT_D, t = i - r * NT_D;
    out_attn[(size_t)(row0 + r) * NT_D + t] = ATW[r * 8 + t];
  }
  // ---- 7) context = ctx @ out_proj^T + out_b -> XCAT[:,64:128]
  {
    v16h a[2];
#pragma unroll
    for (int kc = 0; kc < 2; ++kc) a[kc] = afrag_l(QB, 72, kc * 32, lane);
#pragma unroll
    for (int nt = 0; nt < 4; ++nt) {
      v8f acc = {};
#pragma unroll
      for (int kc = 0; kc < 2; ++kc)
        acc = wmma16(a[kc], bfrag_w(wf + WOFF_OUTPJ, 64, nt * 16, kc * 32, lane), acc);
      store_tile(acc, out_b, nt * 16, XCAT, 264, 64 + nt * 16, lane, false);
    }
  }
  // ---- 8) GRU: r,z via one K=256 GEMM on x'=[h_obs|context|belief]; n-gate split
  for (int nt = 0; nt < 8; ++nt) {
    v8f ar = {}, az = {}, ai = {}, ah = {};
#pragma unroll
    for (int kc = 0; kc < 8; ++kc) {
      v16h a = afrag_l(XCAT, 264, kc * 32, lane);
      ar = wmma16(a, bfrag_w(wf + WOFF_WRZ, 256, nt * 16, kc * 32, lane), ar);
      az = wmma16(a, bfrag_w(wf + WOFF_WRZ, 256, 128 + nt * 16, kc * 32, lane), az);
    }
#pragma unroll
    for (int kc = 0; kc < 4; ++kc) {
      v16h ax = afrag_l(XCAT, 264, kc * 32, lane);        // x part (cols 0..127)
      ai = wmma16(ax, bfrag_w(wf + WOFF_WNI, 128, nt * 16, kc * 32, lane), ai);
      v16h ab = afrag_l(XCAT + 128, 264, kc * 32, lane);  // belief part
      ah = wmma16(ab, bfrag_w(wf + WOFF_WNH, 128, nt * 16, kc * 32, lane), ah);
    }
    int n = lane & 15, mb = (lane >> 4) * 8;
    int col = nt * 16 + n;
    float br  = bih[col] + bhh[col];
    float bz  = bih[128 + col] + bhh[128 + col];
    float bi2 = bih[256 + col], bh2 = bhh[256 + col];
#pragma unroll
    for (int r = 0; r < 8; ++r) {
      int row = row0 + mb + r;
      float rg = 1.f / (1.f + expf(-(ar[r] + br)));
      float zg = 1.f / (1.f + expf(-(az[r] + bz)));
      float ng = tanhf(ai[r] + bi2 + rg * (ah[r] + bh2));
      float bo = pbel[(size_t)row * H_D + col];
      float be = (1.f - zg) * ng + zg * bo;
      out_belief[(size_t)row * H_D + col] = be;
      BEL2[(mb + r) * 136 + col] = (_Float16)be;
    }
  }
  // ---- 9) policy head: logits = belief @ Wpol^T + bpol          [16x128]->[16x16]
  {
    v8f acc = {};
#pragma unroll
    for (int kc = 0; kc < 4; ++kc)
      acc = wmma16(afrag_l(BEL2, 136, kc * 32, lane),
                   bfrag_w(wf + WOFF_WPOL, 128, 0, kc * 32, lane), acc);
    int n = lane & 15, mb = (lane >> 4) * 8;
    float bb = bpol[n];
#pragma unroll
    for (int r = 0; r < 8; ++r)
      out_logits[(size_t)(row0 + mb + r) * A_D + n] = acc[r] + bb;
  }
}

extern "C" void kernel_launch(void* const* d_in, const int* in_sizes, int n_in,
                              void* d_out, int out_size, void* d_ws, size_t ws_size,
                              hipStream_t stream) {
  (void)in_sizes; (void)n_in; (void)out_size; (void)ws_size;
  const float* obs  = (const float*)d_in[0];
  const float* pbel = (const float*)d_in[1];
  const float* tmh  = (const float*)d_in[2];
  const float* W1   = (const float*)d_in[3];
  const float* b1   = (const float*)d_in[4];
  const float* W2   = (const float*)d_in[5];
  const float* b2   = (const float*)d_in[6];
  const float* Wp   = (const float*)d_in[7];
  const float* bp   = (const float*)d_in[8];
  const float* inw  = (const float*)d_in[9];
  const float* inb  = (const float*)d_in[10];
  const float* outw = (const float*)d_in[11];
  const float* outb = (const float*)d_in[12];
  const float* wih  = (const float*)d_in[13];
  const float* whh  = (const float*)d_in[14];
  const float* bih  = (const float*)d_in[15];
  const float* bhh  = (const float*)d_in[16];
  const float* Wpol = (const float*)d_in[17];
  const float* bpol = (const float*)d_in[18];

  _Float16* wf = (_Float16*)d_ws;                  // 274 KB f16 weights
  float* out_logits = (float*)d_out;
  float* out_belief = out_logits + (size_t)B_TOT * A_D;
  float* out_attn   = out_belief + (size_t)B_TOT * H_D;

  pack_weights<<<(W_TOTAL + 255) / 256, 256, 0, stream>>>(
      W1, W2, Wp, inw, outw, wih, whh, Wpol, wf);
  aerial_fused<<<B_TOT / 16, 32, 0, stream>>>(
      obs, pbel, tmh, b1, b2, bp, inb, outb, bih, bhh, bpol, wf,
      out_logits, out_belief, out_attn);
}
